// LNN_27968827031558
// MI455X (gfx1250) — matrix operand order repaired and checked by reading, not compiled, not run on hardware
//
#include <hip/hip_runtime.h>
#include <hip/hip_bf16.h>

// ---------------------------------------------------------------------------
// CfC liquid-NN scan for MI455X (gfx1250).
//  - 8 independent recurrence chains (batch tiles of 16) -> 8 persistent WGs
//  - all GEMMs via v_wmma_f32_16x16x32_bf16 (f32 accumulate)
//  - weights pre-swizzled once into bf16 WMMA B-fragments in d_ws (768 KB,
//    L2-resident); per-lane slice = 32 contiguous bytes
//  - weight loads in saddr+voffset+imm form: SGPR base (opaqued 32-bit scalar
//    offset per step blocks LICM/spilling), lane*32 VGPR offset, kt and
//    inter-matrix deltas in the 24-bit immediate -> zero per-frag addr math
//  - double-buffered kt pipeline: next fragments in flight during each WMMA
//  - activations via v_exp_f32 + v_rcp_f32 (no slow f32 division)
// ---------------------------------------------------------------------------

typedef __attribute__((ext_vector_type(16))) __bf16 v16bf;
typedef __attribute__((ext_vector_type(8)))  __bf16 v8bf;
typedef __attribute__((ext_vector_type(4)))  __bf16 v4bf;
typedef __attribute__((ext_vector_type(8)))  float  v8f;

typedef __attribute__((address_space(1))) const char gchar;
typedef __attribute__((address_space(1))) const v8bf gv8bf;

#define B_    128
#define S_    1024
#define FIN   128
#define H_    256
#define FOUT  128
#define BB_   256

#define WB_KT 12
#define WB_NT 16
#define HD_KT 8
#define HD_NT 16
#define WO_KT 8
#define WO_NT 8

#define FR_WB   0
#define FR_FF1  (FR_WB  + WB_KT * WB_NT)   // 192
#define FR_FF2  (FR_FF1 + HD_KT * HD_NT)   // 320
#define FR_TA   (FR_FF2 + HD_KT * HD_NT)   // 448
#define FR_TB   (FR_TA  + HD_KT * HD_NT)   // 576
#define FR_WO   (FR_TB  + HD_KT * HD_NT)   // 704
#define FR_TOT  (FR_WO  + WO_KT * WO_NT)   // 768 frags * 1KB = 768 KB in d_ws

// byte deltas between head matrices (folded into load immediates)
#define D_FF2 ((FR_FF2 - FR_FF1) * 1024)   // 128 KB
#define D_TA  ((FR_TA  - FR_FF1) * 1024)   // 256 KB
#define D_TB  ((FR_TB  - FR_FF1) * 1024)   // 384 KB

#define AP 392   // abuf row pitch (bf16), 784B rows, 16B aligned
#define ZP 264   // zbuf row pitch (bf16), 528B rows, 16B aligned

union Frag {
    v16bf v;
    v8bf  h[2];
};

static __device__ __forceinline__ v8f wmma_bf16(const Frag& a, const Frag& b, v8f c) {
    return __builtin_amdgcn_wmma_f32_16x16x32_bf16(
        false, a.v, false, b.v, (short)0, c, false, false);
}

// Opaque a wave-uniform scalar byte-offset (SGPR) so the optimizer cannot
// hoist/CSE the dependent weight loads across the 1024-step loop.
static __device__ __forceinline__ unsigned opaque_s(unsigned v) {
    asm volatile("" : "+s"(v));
    return v;
}
static __device__ __forceinline__ unsigned rfl(unsigned v) {
    return __builtin_amdgcn_readfirstlane(v);
}

// Weight fragment load: 2x global_load_b128, saddr + voffset + immediate.
static __device__ __forceinline__ void load_w(Frag& w, const __bf16* frags,
                                              unsigned soff, unsigned voff, int imm) {
    gchar* p = (gchar*)(unsigned long long)frags + soff + voff + imm;
    w.h[0] = *(gv8bf*)(p);
    w.h[1] = *(gv8bf*)(p + 16);
}
// A fragment load from LDS: 2x ds_load_b128.
static __device__ __forceinline__ void load_a(Frag& a, const __bf16* ap) {
    a.h[0] = *(const v8bf*)(ap);
    a.h[1] = *(const v8bf*)(ap + 16);
}

static __device__ __forceinline__ float fast_tanh(float x) {
    float e = __expf(2.0f * x);
    return (e - 1.0f) * __builtin_amdgcn_rcpf(e + 1.0f);
}
static __device__ __forceinline__ float fast_sigmoid(float x) {
    return __builtin_amdgcn_rcpf(1.0f + __expf(-x));
}

// ---------------------------------------------------------------------------
// Kernel 1: pack fp32 row-major weights [K][N] into bf16 WMMA B-fragments.
// lane<16: column N=nt*16+lane, K=kt*32+{0..7,16..23} ; lane>=16: same col set, K+8.
// Stored lane-major: frag*512 + lane*16 bf16 (32 contiguous bytes per lane).
// ---------------------------------------------------------------------------
__global__ void lnn_pack_weights(const float* __restrict__ Wb,
                                 const float* __restrict__ Wff1,
                                 const float* __restrict__ Wff2,
                                 const float* __restrict__ Wta,
                                 const float* __restrict__ Wtb,
                                 const float* __restrict__ Wout,
                                 __bf16* __restrict__ frags)
{
    int f = blockIdx.x;
    int l = threadIdx.x;
    const float* W;
    int N, kt, nt;
    if (f < FR_FF1)      { W = Wb;   N = BB_;  int r = f;          nt = r / WB_KT; kt = r % WB_KT; }
    else if (f < FR_FF2) { W = Wff1; N = H_;   int r = f - FR_FF1; nt = r / HD_KT; kt = r % HD_KT; }
    else if (f < FR_TA)  { W = Wff2; N = H_;   int r = f - FR_FF2; nt = r / HD_KT; kt = r % HD_KT; }
    else if (f < FR_TB)  { W = Wta;  N = H_;   int r = f - FR_TA;  nt = r / HD_KT; kt = r % HD_KT; }
    else if (f < FR_WO)  { W = Wtb;  N = H_;   int r = f - FR_TB;  nt = r / HD_KT; kt = r % HD_KT; }
    else                 { W = Wout; N = FOUT; int r = f - FR_WO;  nt = r / WO_KT; kt = r % WO_KT; }

    int n  = nt * 16 + (l & 15);
    int k0 = kt * 32 + ((l >> 4) << 3);
    __bf16* dst = frags + (size_t)f * 512 + (size_t)l * 16;
#pragma unroll
    for (int j = 0; j < 16; ++j)
        dst[j] = (__bf16)W[(size_t)(k0 + j + (j >= 8 ? 8 : 0)) * N + n];
}

// ---------------------------------------------------------------------------
// Kernel 2: persistent CfC scan. One block per 16-row batch tile, 16 waves.
// ---------------------------------------------------------------------------
__launch_bounds__(512)
__global__ void lnn_cfc_scan(const float* __restrict__ x,    // [B,S,FIN]
                             const float* __restrict__ ts,   // [B,S]
                             const float* __restrict__ bb,
                             const float* __restrict__ bff1,
                             const float* __restrict__ bff2,
                             const float* __restrict__ bta,
                             const float* __restrict__ btb,
                             const float* __restrict__ bout,
                             const __bf16* __restrict__ frags,
                             float* __restrict__ out)        // [B,S,FOUT]
{
    __shared__ __bf16 abuf[16 * AP];   // [x_t (0..127) | h (128..383)] bf16
    __shared__ __bf16 zbuf[16 * ZP];   // backbone output z
    __shared__ float  tsb[16];

    const int tid  = threadIdx.x;
    const int lane = tid & 31;
    const int wave = tid >> 5;       // 0..15
    const int b0   = blockIdx.x * 16;
    const int cl   = lane & 15;      // tile column (B/C)
    const int row  = lane & 15;      // tile row (A)
    const int sel  = lane >> 4;      // K-half (A) / M-half (C) select

    // h0 = 0
    for (int i = tid; i < 16 * H_; i += 512) {
        int m = i >> 8, k = i & 255;
        abuf[m * AP + FIN + k] = (__bf16)0.0f;
    }

    // per-lane biases, fixed for the whole scan
    const float bbv = bb  [wave * 16 + cl];
    const float b1v = bff1[wave * 16 + cl];
    const float b2v = bff2[wave * 16 + cl];
    const float bav = bta [wave * 16 + cl];
    const float btv = btb [wave * 16 + cl];
    const float bov = bout[(wave & 7) * 16 + cl];

    // wave-uniform scalar byte offsets into the fragment arena
    const unsigned s1off0 = rfl((unsigned)((FR_WB + wave * WB_KT) * 1024));
    const unsigned s2off0 = rfl((unsigned)((FR_FF1 + wave * HD_KT) * 1024));
    const unsigned s3off0 = rfl((unsigned)((FR_WO + (wave & 7) * WO_KT) * 1024));
    const unsigned voff   = (unsigned)lane * 32u;   // per-lane slice

    for (int s = 0; s < S_; ++s) {
        // ---- stage 0: load x_t tile (fp32 -> bf16, packed b64 store) + ts --
        {
            int m = tid >> 5;
            int k = (tid & 31) * 4;
            const float4 xv = *reinterpret_cast<const float4*>(
                &x[(((size_t)(b0 + m)) * S_ + s) * FIN + k]);
            v4bf p = { (__bf16)xv.x, (__bf16)xv.y, (__bf16)xv.z, (__bf16)xv.w };
            *reinterpret_cast<v4bf*>(&abuf[m * AP + k]) = p;
            if (tid < 16) tsb[tid] = ts[(size_t)(b0 + tid) * S_ + s];
        }
        __syncthreads();

        // ---- stage 1: z = tanh([x_t,h] @ Wb + bb) ; wave -> N-tile ---------
        {
            const unsigned soff = opaque_s(s1off0);   // re-opaqued each step
            const __bf16* ab = &abuf[row * AP + sel * 8];
            Frag ac, wc, an, wn;
            load_a(ac, ab);
            load_w(wc, frags, soff, voff, 0);
            v8f acc = {};
#pragma unroll
            for (int kt = 0; kt < WB_KT; ++kt) {
                if (kt + 1 < WB_KT) {             // prefetch next kt
                    load_a(an, ab + (kt + 1) * 32);
                    load_w(wn, frags, soff, voff, (kt + 1) * 1024);
                }
                acc = wmma_bf16(ac, wc, acc);
                ac = an; wc = wn;
            }
#pragma unroll
            for (int r = 0; r < 8; ++r) {
                int m = r + sel * 8;
                zbuf[m * ZP + wave * 16 + cl] = (__bf16)fast_tanh(acc[r] + bbv);
            }
        }
        __syncthreads();

        // ---- stage 2: four heads + CfC gate ; wave -> N-tile (all 4 mats) --
        {
            const unsigned soff = opaque_s(s2off0);   // one scalar for 4 mats
            const __bf16* zb = &zbuf[row * ZP + sel * 8];
            Frag ac, w1c, w2c, wac, wbc;
            Frag an, w1n, w2n, wan, wbn;
            load_a(ac, zb);
            load_w(w1c, frags, soff, voff, 0);
            load_w(w2c, frags, soff, voff, D_FF2);
            load_w(wac, frags, soff, voff, D_TA);
            load_w(wbc, frags, soff, voff, D_TB);
            v8f a1 = {}, a2 = {}, aa = {}, abk = {};
#pragma unroll
            for (int kt = 0; kt < HD_KT; ++kt) {
                if (kt + 1 < HD_KT) {             // prefetch next kt
                    int ko = (kt + 1) * 1024;
                    load_a(an, zb + (kt + 1) * 32);
                    load_w(w1n, frags, soff, voff, ko);
                    load_w(w2n, frags, soff, voff, D_FF2 + ko);
                    load_w(wan, frags, soff, voff, D_TA + ko);
                    load_w(wbn, frags, soff, voff, D_TB + ko);
                }
                a1  = wmma_bf16(ac, w1c, a1);
                a2  = wmma_bf16(ac, w2c, a2);
                aa  = wmma_bf16(ac, wac, aa);
                abk = wmma_bf16(ac, wbc, abk);
                ac = an; w1c = w1n; w2c = w2n; wac = wan; wbc = wbn;
            }
            // ts broadcast: each lane needs rows sel*8..sel*8+7 (contiguous)
            float4 t0 = *reinterpret_cast<const float4*>(&tsb[sel * 8]);
            float4 t1 = *reinterpret_cast<const float4*>(&tsb[sel * 8 + 4]);
            float ttv[8] = { t0.x, t0.y, t0.z, t0.w, t1.x, t1.y, t1.z, t1.w };
#pragma unroll
            for (int r = 0; r < 8; ++r) {
                int m = r + sel * 8;
                float ff1 = fast_tanh(a1[r] + b1v);
                float ff2 = fast_tanh(a2[r] + b2v);
                float g   = fast_sigmoid((aa[r] + bav) * ttv[r] + (abk[r] + btv));
                float hn  = ff1 * (1.0f - g) + g * ff2;
                abuf[m * AP + FIN + wave * 16 + cl] = (__bf16)hn;
            }
        }
        __syncthreads();

        // ---- stage 3: y = h_new @ Wout + bout ; waves 0..7 -----------------
        if (wave < 8) {
            const unsigned soff = opaque_s(s3off0);
            const __bf16* hb = &abuf[row * AP + FIN + sel * 8];
            Frag ac, wc, an, wn;
            load_a(ac, hb);
            load_w(wc, frags, soff, voff, 0);
            v8f acc = {};
#pragma unroll
            for (int kt = 0; kt < WO_KT; ++kt) {
                if (kt + 1 < WO_KT) {
                    load_a(an, hb + (kt + 1) * 32);
                    load_w(wn, frags, soff, voff, (kt + 1) * 1024);
                }
                acc = wmma_bf16(ac, wc, acc);
                ac = an; wc = wn;
            }
#pragma unroll
            for (int r = 0; r < 8; ++r) {
                int m = r + sel * 8;
                out[(((size_t)(b0 + m)) * S_ + s) * FOUT + wave * 16 + cl] = acc[r] + bov;
            }
        }
        // no 4th barrier: next stage-0 writes only abuf x-region / tsb, whose
        // last readers completed before the stage-2 barrier.
    }
}

// ---------------------------------------------------------------------------
extern "C" void kernel_launch(void* const* d_in, const int* in_sizes, int n_in,
                              void* d_out, int out_size, void* d_ws, size_t ws_size,
                              hipStream_t stream) {
    (void)in_sizes; (void)n_in; (void)out_size; (void)ws_size;
    const float* x    = (const float*)d_in[0];
    const float* ts   = (const float*)d_in[1];
    const float* Wb   = (const float*)d_in[2];
    const float* bb   = (const float*)d_in[3];
    const float* Wff1 = (const float*)d_in[4];
    const float* bff1 = (const float*)d_in[5];
    const float* Wff2 = (const float*)d_in[6];
    const float* bff2 = (const float*)d_in[7];
    const float* Wta  = (const float*)d_in[8];
    const float* bta  = (const float*)d_in[9];
    const float* Wtb  = (const float*)d_in[10];
    const float* btb  = (const float*)d_in[11];
    const float* Wout = (const float*)d_in[12];
    const float* bout = (const float*)d_in[13];
    __bf16* frags = (__bf16*)d_ws;   // needs 768 KB of workspace

    lnn_pack_weights<<<FR_TOT, 32, 0, stream>>>(Wb, Wff1, Wff2, Wta, Wtb, Wout, frags);
    lnn_cfc_scan<<<B_ / 16, 512, 0, stream>>>(x, ts, bb, bff1, bff2, bta, btb, bout,
                                              frags, (float*)d_out);
}
